// HybridGenerator_20633022890071
// MI455X (gfx1250) — matrix operand order repaired and checked by reading, not compiled, not run on hardware
//
#include <hip/hip_runtime.h>

// ---------- vector types ----------
typedef __attribute__((ext_vector_type(16))) _Float16 v16h;
typedef __attribute__((ext_vector_type(8)))  _Float16 v8h;
typedef __attribute__((ext_vector_type(8)))  float    v8f;

union FragU { v16h v; v8h h[2]; };

// =====================================================================
// Kernel 1: fused front-end + 4-qubit statevector sim -> h2 (fp16)
//   per row:  in8 = [x, emb[cid]];  h = relu(in8 @ W1^T + b1) (fused)
//   angles = h @ W2^T + b2;  z = quantum(angles, qw);
//   h2 = relu(z @ W3^T + b3)  -> stored as _Float16 [B,512]
// =====================================================================

__device__ __forceinline__ void apply1q(float* sr, float* si, int bp,
    float g00r, float g00i, float g01r, float g01i,
    float g10r, float g10i, float g11r, float g11i)
{
#pragma unroll
    for (int i = 0; i < 16; ++i) {
        if (i & (1 << bp)) continue;
        const int j = i | (1 << bp);
        const float ar = sr[i], ai = si[i], br = sr[j], bi = si[j];
        sr[i] = g00r * ar - g00i * ai + g01r * br - g01i * bi;
        si[i] = g00r * ai + g00i * ar + g01r * bi + g01i * br;
        sr[j] = g10r * ar - g10i * ai + g11r * br - g11i * bi;
        si[j] = g10r * ai + g10i * ar + g11r * bi + g11i * br;
    }
}

__global__ void frontend_kernel(
    const float* __restrict__ x, const int* __restrict__ cid,
    const float* __restrict__ emb,
    const float* __restrict__ W1, const float* __restrict__ b1,
    const float* __restrict__ W2, const float* __restrict__ b2,
    const float* __restrict__ qw,
    const float* __restrict__ W3, const float* __restrict__ b3,
    _Float16* __restrict__ h2h, int B)
{
    const int b = blockIdx.x * blockDim.x + threadIdx.x;
    if (b >= B) return;

    // ---- concat input ----
    float in8[8];
    const int c = cid[b];
#pragma unroll
    for (int i = 0; i < 4; ++i) in8[i]     = x[(size_t)b * 4 + i];
#pragma unroll
    for (int i = 0; i < 4; ++i) in8[4 + i] = emb[(size_t)c * 4 + i];

    // ---- fused layer1 (relu) + layer2: accumulate angles, no h[128] ----
    float ang[4] = { b2[0], b2[1], b2[2], b2[3] };
    for (int j = 0; j < 128; ++j) {
        float hj = b1[j];
#pragma unroll
        for (int i = 0; i < 8; ++i) hj += W1[j * 8 + i] * in8[i];
        hj = fmaxf(hj, 0.0f);
#pragma unroll
        for (int w = 0; w < 4; ++w) ang[w] += W2[w * 128 + j] * hj;
    }

    // ---- 4-qubit statevector in registers ----
    float sr[16], si[16];
#pragma unroll
    for (int i = 0; i < 16; ++i) { sr[i] = 0.0f; si[i] = 0.0f; }
    sr[0] = 1.0f;

    // AngleEmbedding: RX(ang[w]) on wire w (bit position 3-w)
#pragma unroll
    for (int w = 0; w < 4; ++w) {
        const float ch = cosf(0.5f * ang[w]);
        const float sh = sinf(0.5f * ang[w]);
        // [[c, -i s], [-i s, c]]
        apply1q(sr, si, 3 - w, ch, 0.0f, 0.0f, -sh, 0.0f, -sh, ch, 0.0f);
    }

    // StronglyEntanglingLayers: 3 layers of Rot + CNOT ring (ranges 1,2,3)
#pragma unroll
    for (int l = 0; l < 3; ++l) {
#pragma unroll
        for (int w = 0; w < 4; ++w) {
            const float phi   = qw[(l * 4 + w) * 3 + 0];
            const float theta = qw[(l * 4 + w) * 3 + 1];
            const float omg   = qw[(l * 4 + w) * 3 + 2];
            const float ct = cosf(0.5f * theta), st = sinf(0.5f * theta);
            const float ap = 0.5f * (phi + omg), am = 0.5f * (phi - omg);
            const float cap = cosf(ap), sap = sinf(ap);
            const float cam = cosf(am), sam = sinf(am);
            // g00 = e^{-i ap} c ; g01 = -e^{+i am} s ; g10 = e^{-i am} s ; g11 = e^{+i ap} c
            apply1q(sr, si, 3 - w,
                    cap * ct, -sap * ct,
                   -cam * st, -sam * st,
                    cam * st, -sam * st,
                    cap * ct,  sap * ct);
        }
#pragma unroll
        for (int w = 0; w < 4; ++w) {
            const int t  = (w + (l % 3) + 1) % 4;
            const int bc = 3 - w, bt = 3 - t;
#pragma unroll
            for (int i = 0; i < 16; ++i) {
                if (!(i & (1 << bc))) continue;
                if (i & (1 << bt)) continue;
                const int j = i | (1 << bt);
                float tr = sr[i]; sr[i] = sr[j]; sr[j] = tr;
                float ti = si[i]; si[i] = si[j]; si[j] = ti;
            }
        }
    }

    // ---- PauliZ expectations ----
    float z[4] = { 0.0f, 0.0f, 0.0f, 0.0f };
#pragma unroll
    for (int i = 0; i < 16; ++i) {
        const float p = sr[i] * sr[i] + si[i] * si[i];
#pragma unroll
        for (int w = 0; w < 4; ++w)
            z[w] += ((i >> (3 - w)) & 1) ? -p : p;
    }

    // ---- layer3: h2 = relu(z @ W3^T + b3), store fp16 ----
    _Float16* dst = h2h + (size_t)b * 512;
    for (int k = 0; k < 512; ++k) {
        float v = b3[k];
#pragma unroll
        for (int j = 0; j < 4; ++j) v += W3[k * 4 + j] * z[j];
        v = fmaxf(v, 0.0f);
        dst[k] = (_Float16)v;
    }
}

// =====================================================================
// Kernel 2: W4 fp32 -> fp16 (keeps [4096][512] row-major layout)
// =====================================================================
__global__ void cvt_w4_kernel(const float* __restrict__ W4,
                              _Float16* __restrict__ W4h, int n)
{
    int i = (blockIdx.x * blockDim.x + threadIdx.x) * 4;
    if (i >= n) return;
    const float4 v = *reinterpret_cast<const float4*>(W4 + i);
    W4h[i + 0] = (_Float16)v.x;
    W4h[i + 1] = (_Float16)v.y;
    W4h[i + 2] = (_Float16)v.z;
    W4h[i + 3] = (_Float16)v.w;
}

// =====================================================================
// Kernel 3: WMMA GEMM  out = tanh(h2[16384,512] @ W4h[4096,512]^T + b4)
//   block tile 128x128, 8 waves as 2(M) x 4(N), each wave 64x32
//   K staged through LDS in 64-wide slabs, v_wmma_f32_16x16x32_f16
// =====================================================================
#define LDSTR 72   // padded LDS row stride in halves (avoids bank conflicts)

__global__ __launch_bounds__(256) void gemm_tanh_kernel(
    const _Float16* __restrict__ A,   // h2h  [B,512]
    const _Float16* __restrict__ Bw,  // W4h  [4096,512]
    const float* __restrict__ bias,   // b4   [4096]
    float* __restrict__ out, int Brows)
{
    __shared__ _Float16 As[128 * LDSTR];
    __shared__ _Float16 Bs[128 * LDSTR];

    const int t    = threadIdx.x;
    const int nblk = blockIdx.x;   // 0..31   (N / 128)
    const int mblk = blockIdx.y;   // 0..127  (M / 128)
    const int lane = t & 31;
    const int wv   = t >> 5;       // 0..7
    const int wm   = wv >> 2;      // 0..1  (M position of wave)
    const int wn   = wv & 3;       // 0..3  (N position of wave)
    const int lr   = lane & 15;
    const int hi   = lane >> 4;

    v8f acc[4][2];
#pragma unroll
    for (int mi = 0; mi < 4; ++mi)
#pragma unroll
        for (int ni = 0; ni < 2; ++ni) acc[mi][ni] = 0.0f;

    // cooperative stage-load mapping: thread t loads 32 halves of one row
    const int lrow = t >> 1;            // 0..127
    const int lk   = (t & 1) * 32;      // 0 or 32 halves
    const _Float16* gA = A  + (size_t)(mblk * 128 + lrow) * 512 + lk;
    const _Float16* gB = Bw + (size_t)(nblk * 128 + lrow) * 512 + lk;
    uint4* da = reinterpret_cast<uint4*>(&As[lrow * LDSTR + lk]);
    uint4* db = reinterpret_cast<uint4*>(&Bs[lrow * LDSTR + lk]);

    for (int kb = 0; kb < 512; kb += 64) {
        const uint4* sa = reinterpret_cast<const uint4*>(gA + kb);
        const uint4* sb = reinterpret_cast<const uint4*>(gB + kb);
#pragma unroll
        for (int q = 0; q < 4; ++q) { da[q] = sa[q]; db[q] = sb[q]; }
        if (kb + 64 < 512) {           // global_prefetch next K slab
            __builtin_prefetch(gA + kb + 64, 0, 0);
            __builtin_prefetch(gB + kb + 64, 0, 0);
        }
        __syncthreads();

#pragma unroll
        for (int k32 = 0; k32 < 64; k32 += 32) {
            // A fragments: 16-bit A 16x32 layout — lane holds row m=lr,
            // halves 0..7 = K (hi*8 .. +7), halves 8..15 = K 16+(hi*8) ..
            FragU af[4];
#pragma unroll
            for (int mi = 0; mi < 4; ++mi) {
                const int row = wm * 64 + mi * 16 + lr;
                af[mi].h[0] = *reinterpret_cast<const v8h*>(
                    &As[row * LDSTR + k32 + hi * 8]);
                af[mi].h[1] = *reinterpret_cast<const v8h*>(
                    &As[row * LDSTR + k32 + 16 + hi * 8]);
            }
            // B fragments: lane holds col n=lr; halves 0..7 = K hi*8.., 8..15 = K 16+hi*8..
            FragU bf[2];
#pragma unroll
            for (int ni = 0; ni < 2; ++ni) {
                const int col = wn * 32 + ni * 16 + lr;
                const v8h* p = reinterpret_cast<const v8h*>(
                    &Bs[col * LDSTR + k32 + hi * 8]);
                bf[ni].h[0] = p[0];
                bf[ni].h[1] = p[2];
            }
#pragma unroll
            for (int mi = 0; mi < 4; ++mi)
#pragma unroll
                for (int ni = 0; ni < 2; ++ni)
                    acc[mi][ni] = __builtin_amdgcn_wmma_f32_16x16x32_f16(
                        false, af[mi].v, false, bf[ni].v,
                        (short)0, acc[mi][ni], false, false);
        }
        __syncthreads();
    }

    // epilogue: bias + tanh, fp32 out.  C/D layout: lane n=lr,
    // vgpr r holds row m = r + 8*hi within the 16x16 fragment.
#pragma unroll
    for (int ni = 0; ni < 2; ++ni) {
        const int n  = nblk * 128 + wn * 32 + ni * 16 + lr;
        const float bv = bias[n];
#pragma unroll
        for (int mi = 0; mi < 4; ++mi) {
#pragma unroll
            for (int r = 0; r < 8; ++r) {
                const int m = mblk * 128 + wm * 64 + mi * 16 + hi * 8 + r;
                out[(size_t)m * 4096 + n] = tanhf(acc[mi][ni][r] + bv);
            }
        }
    }
}

// =====================================================================
// launch
// =====================================================================
extern "C" void kernel_launch(void* const* d_in, const int* in_sizes, int n_in,
                              void* d_out, int out_size, void* d_ws, size_t ws_size,
                              hipStream_t stream) {
    const float* x   = (const float*)d_in[0];
    const int*   cid = (const int*)  d_in[1];
    const float* emb = (const float*)d_in[2];
    const float* W1  = (const float*)d_in[3];
    const float* b1  = (const float*)d_in[4];
    const float* W2  = (const float*)d_in[5];
    const float* b2  = (const float*)d_in[6];
    const float* qw  = (const float*)d_in[7];
    const float* W3  = (const float*)d_in[8];
    const float* b3  = (const float*)d_in[9];
    const float* W4  = (const float*)d_in[10];
    const float* b4  = (const float*)d_in[11];
    float* out = (float*)d_out;

    const int B = in_sizes[0] / 4;   // 16384

    _Float16* h2h = (_Float16*)d_ws;                          // B*512 halves
    _Float16* W4h = (_Float16*)((char*)d_ws +
                                (size_t)B * 512 * sizeof(_Float16));

    // front-end + quantum sim -> h2 (fp16)
    frontend_kernel<<<(B + 255) / 256, 256, 0, stream>>>(
        x, cid, emb, W1, b1, W2, b2, qw, W3, b3, h2h, B);

    // W4 -> fp16
    const int nW4 = 4096 * 512;
    cvt_w4_kernel<<<(nW4 / 4 + 255) / 256, 256, 0, stream>>>(W4, W4h, nW4);

    // final WMMA GEMM + bias + tanh
    dim3 grid(4096 / 128, B / 128);
    gemm_tanh_kernel<<<grid, 256, 0, stream>>>(h2h, W4h, b4, out, B);
}
